// PRPN_58566174048316
// MI455X (gfx1250) — matrix operand order repaired and checked by reading, not compiled, not run on hardware
//
#include <hip/hip_runtime.h>
#include <math.h>

#define T_N 192
#define B_N 32
#define D_N 384
#define H_N 384
#define S_N 15
#define NL_N 2
#define G4_N 1536
#define BN_SC 0.9999950000374997f
#define INV_SQRT_H 0.05103103630798287f

typedef __attribute__((ext_vector_type(16))) _Float16 v16h;
typedef __attribute__((ext_vector_type(8))) float v8f;

__device__ __forceinline__ v8f wmma16(v16h a, v16h b, v8f c) {
  return __builtin_amdgcn_wmma_f32_16x16x32_f16(false, a, false, b, (short)0, c,
                                                false, false);
}

__device__ __forceinline__ float wredsum(float v) {
#pragma unroll
  for (int o = 16; o > 0; o >>= 1) v += __shfl_xor(v, o, 32);
  return v;
}
__device__ __forceinline__ float wredmax(float v) {
#pragma unroll
  for (int o = 16; o > 0; o >>= 1) v = fmaxf(v, __shfl_xor(v, o, 32));
  return v;
}
__device__ __forceinline__ float sigf(float x) { return 1.f / (1.f + expf(-x)); }

// A-fragment scatter index for 16-bit WMMA A tiles (KT k-tiles of 32):
// element (m,k) -> [(mt*KT+kt)*32 + lane]*16 + j
__device__ __forceinline__ int afrag_idx(int m, int k, int KT) {
  int mt = m >> 4, kt = k >> 5, r = k & 31;
  int lane = (m & 15) | (((r >> 3) & 1) << 4);
  int j = (r & 7) | ((r & 16) >> 1);
  return (((mt * KT + kt) << 5) + lane) * 16 + j;
}

// pack 16 f32 (two 8-spans) into a v16h A fragment
__device__ __forceinline__ v16h pack_a(float4 a0, float4 a1, float4 b0,
                                       float4 b1) {
  v16h a;
  a[0] = (_Float16)a0.x; a[1] = (_Float16)a0.y;
  a[2] = (_Float16)a0.z; a[3] = (_Float16)a0.w;
  a[4] = (_Float16)a1.x; a[5] = (_Float16)a1.y;
  a[6] = (_Float16)a1.z; a[7] = (_Float16)a1.w;
  a[8] = (_Float16)b0.x; a[9] = (_Float16)b0.y;
  a[10] = (_Float16)b0.z; a[11] = (_Float16)b0.w;
  a[12] = (_Float16)b1.x; a[13] = (_Float16)b1.y;
  a[14] = (_Float16)b1.z; a[15] = (_Float16)b1.w;
  return a;
}

__device__ __forceinline__ v16h load_a_span(const float* s) {
  float4 x0 = *(const float4*)s;
  float4 x1 = *(const float4*)(s + 4);
  float4 x2 = *(const float4*)(s + 16);
  float4 x3 = *(const float4*)(s + 20);
  return pack_a(x0, x1, x2, x3);
}

// ---------------- weight packing into WMMA B-fragment order ----------------
__global__ __launch_bounds__(256) void pack_w_kernel(
    const float* __restrict__ W, _Float16* __restrict__ out, int Nout, int K,
    int total) {
  int i = blockIdx.x * 256 + threadIdx.x;
  if (i >= total) return;
  int NT = Nout >> 4;
  int tile = i >> 9, r = i & 511;
  int lane = r >> 4, j = r & 15;
  int nt = tile % NT, kt = tile / NT;
  int n = nt * 16 + (lane & 15);
  int k = kt * 32 + ((j & 8) << 1) + ((lane >= 16) ? 8 : 0) + (j & 7);
  out[i] = (_Float16)W[(size_t)n * K + k];
}

// conv weights (H,D,6) -> B[k=l*384+d][n=c]
__global__ __launch_bounds__(256) void pack_conv_kernel(
    const float* __restrict__ W, _Float16* __restrict__ out, int total) {
  int i = blockIdx.x * 256 + threadIdx.x;
  if (i >= total) return;
  int tile = i >> 9, r = i & 511;
  int lane = r >> 4, j = r & 15;
  int nt = tile % 24, kt = tile / 24;
  int n = nt * 16 + (lane & 15);
  int k = kt * 32 + ((j & 8) << 1) + ((lane >= 16) ? 8 : 0) + (j & 7);
  int l = k / D_N, d = k % D_N;
  out[i] = (_Float16)W[((size_t)n * D_N + d) * 6 + l];
}

// ---------------- small utility kernels ----------------
__global__ __launch_bounds__(256) void embed_kernel(const int* __restrict__ ids,
                                                    const float* __restrict__ ew,
                                                    float* __restrict__ emb,
                                                    int total) {
  int i = blockIdx.x * 256 + threadIdx.x;
  if (i >= total) return;
  int m = i / D_N, d = i % D_N;
  emb[i] = ew[(size_t)ids[m] * D_N + d];
}

__global__ __launch_bounds__(256) void zero_kernel(float* __restrict__ p, int n) {
  int i = blockIdx.x * 256 + threadIdx.x;
  if (i < n) p[i] = 0.f;
}

__global__ __launch_bounds__(256) void mask_kernel(const int* __restrict__ ids,
                                                   float* __restrict__ out, int n) {
  int i = blockIdx.x * 256 + threadIdx.x;
  if (i < n) out[i] = (ids[i] != 0) ? 1.f : 0.f;
}

// ---------------- parsing conv as WMMA GEMM ----------------
// rows m = b*192 + t ; tiles never straddle batch boundary (192 = 12*16)
__global__ __launch_bounds__(256) void conv_gemm_kernel(
    const float* __restrict__ emb, const _Float16* __restrict__ wp,
    const float* __restrict__ b1, const float* __restrict__ bng,
    const float* __restrict__ bnb, float* __restrict__ h1) {
  int lane = threadIdx.x & 31, wave = threadIdx.x >> 5;
  int tile = blockIdx.x * 8 + wave;  // 384 mtiles * 24 ntiles
  if (tile >= 9216) return;
  int mt = tile / 24, nt = tile % 24;
  int bb = mt / 12;
  int ttb = (mt % 12) * 16;          // tile-local t base (wave-uniform)
  int tt = ttb + (lane & 15);
  int kbl = (lane >> 4) << 3;
  const _Float16* wpl = wp + nt * 512 + lane * 16;
  v8f acc = {};
  int rowoff = ((tt - 5) * B_N + bb) * D_N + kbl;  // l=0 base (maybe OOB, unread)
  if (ttb != 0) {
    // safe: every tap in range, branch-free immediate-offset loads
    for (int l = 0; l < 6; ++l) {
      const float* rl = emb + rowoff;
#pragma unroll
      for (int dt = 0; dt < 12; ++dt) {
        v16h a = load_a_span(rl + dt * 32);
        v16h b = *(const v16h*)(wpl + (size_t)(l * 12 + dt) * (24 * 512));
        acc = wmma16(a, b, acc);
      }
      rowoff += B_N * D_N;
    }
  } else {
    for (int l = 0; l < 6; ++l) {
      bool ok = (tt + l >= 5);
      const float* rl = emb + rowoff;
#pragma unroll
      for (int dt = 0; dt < 12; ++dt) {
        float4 x0 = {0, 0, 0, 0}, x1 = {0, 0, 0, 0};
        float4 x2 = {0, 0, 0, 0}, x3 = {0, 0, 0, 0};
        if (ok) {
          const float* s = rl + dt * 32;
          x0 = *(const float4*)s;
          x1 = *(const float4*)(s + 4);
          x2 = *(const float4*)(s + 16);
          x3 = *(const float4*)(s + 20);
        }
        v16h a = pack_a(x0, x1, x2, x3);
        v16h b = *(const v16h*)(wpl + (size_t)(l * 12 + dt) * (24 * 512));
        acc = wmma16(a, b, acc);
      }
      rowoff += B_N * D_N;
    }
  }
  int mb = mt * 16 + ((lane >= 16) ? 8 : 0);
  int n = nt * 16 + (lane & 15);
#pragma unroll
  for (int r2 = 0; r2 < 8; ++r2) {
    float y = acc[r2] + b1[n];
    y = bng[n] * y * BN_SC + bnb[n];
    h1[(size_t)(mb + r2) * H_N + n] = fmaxf(y, 0.f);
  }
}

// grouped 1x1 conv -> gate logits -> sigmoid
__global__ __launch_bounds__(256) void gates_kernel(
    const float* __restrict__ h1, const float* __restrict__ w2,
    const float* __restrict__ b2, float* __restrict__ gate,
    float* __restrict__ gaten) {
  int lane = threadIdx.x & 31, wave = threadIdx.x >> 5;
  int m = blockIdx.x * 8 + wave;
  if (m >= B_N * T_N) return;
  float s0 = 0.f, s1 = 0.f;
#pragma unroll
  for (int k = 0; k < 6; ++k) {
    int c = lane + k * 32;
    s0 += h1[(size_t)m * H_N + c] * w2[c];
    s1 += h1[(size_t)m * H_N + 192 + c] * w2[192 + c];
  }
  s0 = wredsum(s0);
  s1 = wredsum(s1);
  if (lane == 0) {
    gate[m] = sigf(s0 + b2[0]);
    gaten[m] = sigf(s1 + b2[1]);
  }
}

// hard memory gates; gate_hat built from `gate` for BOTH mg and mgn (reference)
__global__ __launch_bounds__(256) void memgates_kernel(
    const float* __restrict__ gate, const float* __restrict__ gaten,
    float* __restrict__ mg, float* __restrict__ mgn) {
  int i = blockIdx.x * 256 + threadIdx.x;
  if (i >= B_N * T_N) return;
  int b = i / T_N, t = i % T_N;
  float g = gate[(size_t)b * T_N + t];
  float gn = gaten[(size_t)b * T_N + t];
  float p = 1.f, pn = 1.f;
  for (int s = 0; s < S_N; ++s) {
    float gh = (s > t) ? 1e9f : gate[(size_t)b * T_N + (t - s)];
    float u = (fminf(fmaxf((g - gh) * 20.f + 1.f, -1.f), 1.f) + 1.f) * 0.5f;
    p *= u;
    mg[((size_t)t * B_N + b) * S_N + s] = p;
    float un = (fminf(fmaxf((gn - gh) * 20.f + 1.f, -1.f), 1.f) + 1.f) * 0.5f;
    pn *= un;
    mgn[((size_t)t * B_N + b) * S_N + s] = pn;
  }
}

// ---------------- per-timestep scan kernel ----------------
struct SP {
  const float *emb, *mg, *mgn;
  const float *r_bih, *r_bhh, *r_lnih_g, *r_lnih_b, *r_lnhh_g, *r_lnhh_b;
  const float *r_lnc_g, *r_lnc_b, *r_proj_b, *p_proj_b;
  const _Float16 *wproj, *wih, *whh, *wpproj;
  float *memh, *memc, *pmem, *outh, *outsel;
  float *keyg, *selc, *gih, *ghh, *hglob;
};

__global__ __launch_bounds__(512) void step_kernel(SP p, int t) {
  // h and sel_h kept directly as packed f16 WMMA A-fragments (2 mtiles x 12 kt)
  __shared__ __attribute__((aligned(32))) _Float16 hA[2 * 12 * 512];
  __shared__ __attribute__((aligned(32))) _Float16 selA[2 * 12 * 512];
  __shared__ float attL[B_N * 16];
  __shared__ float attW[B_N * 16];
  const int tid = threadIdx.x;
  const int lane = tid & 31;
  const int wave = tid >> 5;              // 16 waves
  const int eb = tid >> 4;                // 0..31: batch row for elementwise loops
  const int ed = tid & 15;                // 16 threads per row
  const int plast = (t + S_N - 1) % S_N;  // ring slot of logical newest
  const int wslot = t % S_N;              // ring slot to overwrite

  // h0 = emb_t -> hA fragments
  {
    const float* src = p.emb + (size_t)t * B_N * D_N + eb * H_N;
#pragma unroll
    for (int k = 0; k < 24; ++k) {
      int d = ed + (k << 4);
      hA[afrag_idx(eb, d, 12)] = (_Float16)src[d];
    }
  }
  __syncthreads();

  for (int j = 0; j < NL_N; ++j) {
    const float* mh = p.memh + (size_t)j * B_N * S_N * H_N;
    const float* mc = p.memc + (size_t)j * B_N * S_N * H_N;

    // ---- key = concat(h, mh_last) @ r_proj^T + b  (32x384, K=768) ----
    {
      const _Float16* wp = p.wproj + (size_t)j * (24 * 24 * 512);
      const float* bias = p.r_proj_b + j * H_N;
      for (int tile = wave; tile < 48; tile += 16) {
        int mt = tile / 24, nt = tile % 24;
        int row = mt * 16 + (lane & 15);
        int kbl = (lane >> 4) << 3;
        const float* mrow = mh + ((size_t)row * S_N + plast) * H_N + kbl;
        const _Float16* wpl = wp + nt * 512 + lane * 16;
        const _Float16* aF = hA + ((mt * 12) << 9) + lane * 16;
        v8f acc = {};
#pragma unroll
        for (int kt = 0; kt < 12; ++kt) {
          v16h a = *(const v16h*)(aF + (kt << 9));
          v16h b = *(const v16h*)(wpl + (size_t)kt * (24 * 512));
          acc = wmma16(a, b, acc);
        }
#pragma unroll
        for (int kt = 0; kt < 12; ++kt) {
          v16h a = load_a_span(mrow + kt * 32);
          v16h b = *(const v16h*)(wpl + (size_t)(kt + 12) * (24 * 512));
          acc = wmma16(a, b, acc);
        }
        int mb = mt * 16 + ((lane >= 16) ? 8 : 0);
        int n = nt * 16 + (lane & 15);
#pragma unroll
        for (int r2 = 0; r2 < 8; ++r2)
          p.keyg[(size_t)(mb + r2) * H_N + n] = acc[r2] + bias[n];
      }
    }
    __threadfence();
    __syncthreads();

    // ---- attention logits over S slots ----
    for (int b = wave; b < B_N; b += 16) {
      const float* krow = p.keyg + (size_t)b * H_N;
      for (int s = 0; s < S_N; ++s) {
        int phys = (t + s) % S_N;
        const float* mrow = mh + ((size_t)b * S_N + phys) * H_N;
        float sum = 0.f;
#pragma unroll
        for (int k = 0; k < 12; ++k) {
          int d = lane + (k << 5);
          sum += mrow[d] * krow[d];
        }
        sum = wredsum(sum);
        if (lane == 0) attL[b * 16 + s] = sum * INV_SQRT_H;
      }
    }
    __syncthreads();
    // ---- masked softmax (gate = mg_t) ----
    for (int b = wave; b < B_N; b += 16) {
      float l = (lane < S_N) ? attL[b * 16 + lane] : -1e30f;
      float mx = wredmax(l);
      float g = (lane < S_N) ? p.mg[((size_t)t * B_N + b) * S_N + lane] : 0.f;
      float e = (lane < S_N) ? expf(l - mx) * g : 0.f;
      float sm = wredsum(e);
      if (lane < S_N) attW[b * 16 + lane] = e / (sm + 1e-8f);
    }
    __syncthreads();
    // ---- sel_h (-> selA frags) and sel_c ----
    {
      const float* mrowb = mh + (size_t)eb * S_N * H_N;
      const float* mcowb = mc + (size_t)eb * S_N * H_N;
      float* scb = p.selc + eb * H_N;
#pragma unroll
      for (int k = 0; k < 24; ++k) {
        int d = ed + (k << 4);
        float sh = 0.f, sc = 0.f;
#pragma unroll
        for (int s = 0; s < S_N; ++s) {
          int phys = (t + s) % S_N;
          float w = attW[eb * 16 + s];
          sh += w * mrowb[phys * H_N + d];
          sc += w * mcowb[phys * H_N + d];
        }
        selA[afrag_idx(eb, d, 12)] = (_Float16)sh;
        scb[d] = sc;
      }
    }
    __threadfence();
    __syncthreads();

    // ---- gih = h @ Wih^T + bih ; ghh = sel_h @ Whh^T + bhh  (32x1536, K=384) --
    {
      const _Float16* wih_j = p.wih + (size_t)j * (12 * 96 * 512);
      const _Float16* whh_j = p.whh + (size_t)j * (12 * 96 * 512);
      for (int tt2 = wave; tt2 < 384; tt2 += 16) {
        bool ihsel = (tt2 < 192);
        const _Float16* wp = ihsel ? wih_j : whh_j;
        const _Float16* aF = ihsel ? hA : selA;
        const float* bias = ihsel ? (p.r_bih + j * G4_N) : (p.r_bhh + j * G4_N);
        float* outg = ihsel ? p.gih : p.ghh;
        int tile = ihsel ? tt2 : tt2 - 192;
        int mt = tile / 96, nt = tile % 96;
        const _Float16* wpl = wp + nt * 512 + lane * 16;
        const _Float16* aB = aF + ((mt * 12) << 9) + lane * 16;
        v8f acc = {};
#pragma unroll
        for (int kt = 0; kt < 12; ++kt) {
          v16h a = *(const v16h*)(aB + (kt << 9));
          v16h b = *(const v16h*)(wpl + (size_t)kt * (96 * 512));
          acc = wmma16(a, b, acc);
        }
        int mb = mt * 16 + ((lane >= 16) ? 8 : 0);
        int n = nt * 16 + (lane & 15);
#pragma unroll
        for (int r2 = 0; r2 < 8; ++r2)
          outg[(size_t)(mb + r2) * G4_N + n] = acc[r2] + bias[n];
      }
    }
    __threadfence();
    __syncthreads();

    // ---- LayerNorm(ih)+LayerNorm(hh) -> gates -> LSTM cell (wave per row) ----
    for (int b = wave; b < B_N; b += 16) {
      const float* gi = p.gih + (size_t)b * G4_N;
      const float* gh2 = p.ghh + (size_t)b * G4_N;
      float s1 = 0.f, s2 = 0.f, s3 = 0.f, s4 = 0.f;
#pragma unroll
      for (int k = 0; k < 48; ++k) {
        int c = lane + (k << 5);
        float x = gi[c], y = gh2[c];
        s1 += x; s2 += x * x; s3 += y; s4 += y * y;
      }
      s1 = wredsum(s1); s2 = wredsum(s2); s3 = wredsum(s3); s4 = wredsum(s4);
      float mui = s1 * (1.f / 1536.f);
      float ri = 1.f / (sqrtf((s2 - 1536.f * mui * mui) * (1.f / 1535.f)) + 1e-6f);
      float muh = s3 * (1.f / 1536.f);
      float rh = 1.f / (sqrtf((s4 - 1536.f * muh * muh) * (1.f / 1535.f)) + 1e-6f);
      const float* gig = p.r_lnih_g + j * G4_N;
      const float* gib = p.r_lnih_b + j * G4_N;
      const float* ghg = p.r_lnhh_g + j * G4_N;
      const float* ghb = p.r_lnhh_b + j * G4_N;
      float cn[12], on[12];
      float cs = 0.f, css = 0.f;
#pragma unroll
      for (int k2 = 0; k2 < 12; ++k2) {
        int d = lane + 32 * k2;
        int di = d, df = d + 384, dc = d + 768, dz = d + 1152;
        float ig = gig[di] * (gi[di] - mui) * ri + gib[di] +
                   ghg[di] * (gh2[di] - muh) * rh + ghb[di];
        float fg = gig[df] * (gi[df] - mui) * ri + gib[df] +
                   ghg[df] * (gh2[df] - muh) * rh + ghb[df];
        float cg = gig[dc] * (gi[dc] - mui) * ri + gib[dc] +
                   ghg[dc] * (gh2[dc] - muh) * rh + ghb[dc];
        float og = gig[dz] * (gi[dz] - mui) * ri + gib[dz] +
                   ghg[dz] * (gh2[dz] - muh) * rh + ghb[dz];
        float c_new =
            sigf(fg) * p.selc[(size_t)b * H_N + d] + sigf(ig) * tanhf(cg);
        cn[k2] = c_new;
        on[k2] = og;
        cs += c_new;
        css += c_new * c_new;
      }
      cs = wredsum(cs);
      css = wredsum(css);
      float muc = cs * (1.f / 384.f);
      float rc = 1.f / (sqrtf((css - 384.f * muc * muc) * (1.f / 383.f)) + 1e-6f);
      const float* cg2 = p.r_lnc_g + j * H_N;
      const float* cb2 = p.r_lnc_b + j * H_N;
      float* mhw = p.memh + ((size_t)j * B_N * S_N + (size_t)b * S_N + wslot) * H_N;
      float* mcw = p.memc + ((size_t)j * B_N * S_N + (size_t)b * S_N + wslot) * H_N;
#pragma unroll
      for (int k2 = 0; k2 < 12; ++k2) {
        int d = lane + 32 * k2;
        float lnc = cg2[d] * (cn[k2] - muc) * rc + cb2[d];
        float hn = sigf(on[k2]) * tanhf(lnc);
        mcw[d] = cn[k2];
        mhw[d] = hn;
        hA[afrag_idx(b, d, 12)] = (_Float16)hn;
        p.hglob[b * H_N + d] = hn;
        if (j == NL_N - 1) p.outh[((size_t)t * B_N + b) * H_N + d] = hn;
      }
    }
    __threadfence();
    __syncthreads();
  }

  // ---------------- Predict attention ----------------
  for (int tile = wave; tile < 48; tile += 16) {
    int mt = tile / 24, nt = tile % 24;
    int row = mt * 16 + (lane & 15);
    int kbl = (lane >> 4) << 3;
    const float* mrow = p.pmem + ((size_t)row * S_N + plast) * H_N + kbl;
    const _Float16* wpl = p.wpproj + nt * 512 + lane * 16;
    const _Float16* aF = hA + ((mt * 12) << 9) + lane * 16;
    v8f acc = {};
#pragma unroll
    for (int kt = 0; kt < 12; ++kt) {
      v16h a = *(const v16h*)(aF + (kt << 9));
      v16h b = *(const v16h*)(wpl + (size_t)kt * (24 * 512));
      acc = wmma16(a, b, acc);
    }
#pragma unroll
    for (int kt = 0; kt < 12; ++kt) {
      v16h a = load_a_span(mrow + kt * 32);
      v16h b = *(const v16h*)(wpl + (size_t)(kt + 12) * (24 * 512));
      acc = wmma16(a, b, acc);
    }
    int mb = mt * 16 + ((lane >= 16) ? 8 : 0);
    int n = nt * 16 + (lane & 15);
#pragma unroll
    for (int r2 = 0; r2 < 8; ++r2)
      p.keyg[(size_t)(mb + r2) * H_N + n] = acc[r2] + p.p_proj_b[n];
  }
  __threadfence();
  __syncthreads();
  for (int b = wave; b < B_N; b += 16) {
    const float* krow = p.keyg + (size_t)b * H_N;
    for (int s = 0; s < S_N; ++s) {
      int phys = (t + s) % S_N;
      const float* mrow = p.pmem + ((size_t)b * S_N + phys) * H_N;
      float sum = 0.f;
#pragma unroll
      for (int k = 0; k < 12; ++k) {
        int d = lane + (k << 5);
        sum += mrow[d] * krow[d];
      }
      sum = wredsum(sum);
      if (lane == 0) attL[b * 16 + s] = sum * INV_SQRT_H;
    }
  }
  __syncthreads();
  for (int b = wave; b < B_N; b += 16) {
    float l = (lane < S_N) ? attL[b * 16 + lane] : -1e30f;
    float mx = wredmax(l);
    float g = (lane < S_N) ? p.mgn[((size_t)t * B_N + b) * S_N + lane] : 0.f;
    float e = (lane < S_N) ? expf(l - mx) * g : 0.f;
    float sm = wredsum(e);
    if (lane < S_N) attW[b * 16 + lane] = e / (sm + 1e-8f);
  }
  __syncthreads();
  {
    const float* prowb = p.pmem + (size_t)eb * S_N * H_N;
    float* orow = p.outsel + ((size_t)t * B_N + eb) * H_N;
#pragma unroll
    for (int k = 0; k < 24; ++k) {
      int d = ed + (k << 4);
      float s = 0.f;
#pragma unroll
      for (int s2 = 0; s2 < S_N; ++s2) {
        int phys = (t + s2) % S_N;
        s += attW[eb * 16 + s2] * prowb[phys * H_N + d];
      }
      orow[d] = s;
    }
  }
  __syncthreads();  // all pmem reads done before overwrite
  {
    const float* hsrc = p.hglob + eb * H_N;
    float* pdst = p.pmem + ((size_t)eb * S_N + wslot) * H_N;
#pragma unroll
    for (int k = 0; k < 24; ++k) {
      int d = ed + (k << 4);
      pdst[d] = hsrc[d];
    }
  }
}

// ---------------- output head: [out_h|out_sel] @ p_ffd^T, BN, tanh -----------
__global__ __launch_bounds__(256) void head_gemm_kernel(
    const float* __restrict__ outh, const float* __restrict__ outsel,
    const _Float16* __restrict__ wp, const float* __restrict__ bias,
    const float* __restrict__ bg, const float* __restrict__ bb,
    float* __restrict__ out) {
  int lane = threadIdx.x & 31, wave = threadIdx.x >> 5;
  int tile = blockIdx.x * 8 + wave;  // 384*24 tiles
  if (tile >= 9216) return;
  int mt = tile / 24, nt = tile % 24;
  int row = mt * 16 + (lane & 15);
  int kbl = (lane >> 4) << 3;
  const float* rh = outh + (size_t)row * H_N + kbl;
  const float* rs = outsel + (size_t)row * H_N + kbl;
  const _Float16* wpl = wp + nt * 512 + lane * 16;
  v8f acc = {};
#pragma unroll
  for (int kt = 0; kt < 12; ++kt) {
    v16h a = load_a_span(rh + kt * 32);
    v16h b = *(const v16h*)(wpl + (size_t)kt * (24 * 512));
    acc = wmma16(a, b, acc);
  }
#pragma unroll
  for (int kt = 0; kt < 12; ++kt) {
    v16h a = load_a_span(rs + kt * 32);
    v16h b = *(const v16h*)(wpl + (size_t)(kt + 12) * (24 * 512));
    acc = wmma16(a, b, acc);
  }
  int mb = mt * 16 + ((lane >= 16) ? 8 : 0);
  int n = nt * 16 + (lane & 15);
#pragma unroll
  for (int r2 = 0; r2 < 8; ++r2) {
    float y = acc[r2] + bias[n];
    out[(size_t)(mb + r2) * D_N + n] = tanhf(bg[n] * y * BN_SC + bb[n]);
  }
}

// ---------------- host launcher ----------------
extern "C" void kernel_launch(void* const* d_in, const int* in_sizes, int n_in,
                              void* d_out, int out_size, void* d_ws,
                              size_t ws_size, hipStream_t stream) {
  (void)in_sizes; (void)n_in; (void)out_size; (void)ws_size;
  const int* ids = (const int*)d_in[0];
  const float* emb_w = (const float*)d_in[1];
  const float* pconv1_w = (const float*)d_in[2];
  const float* pconv1_b = (const float*)d_in[3];
  const float* pbn_g = (const float*)d_in[4];
  const float* pbn_b = (const float*)d_in[5];
  const float* pconv2_w = (const float*)d_in[6];
  const float* pconv2_b = (const float*)d_in[7];
  const float* r_wih = (const float*)d_in[8];
  const float* r_bih = (const float*)d_in[9];
  const float* r_whh = (const float*)d_in[10];
  const float* r_bhh = (const float*)d_in[11];
  const float* r_lnih_g = (const float*)d_in[12];
  const float* r_lnih_b = (const float*)d_in[13];
  const float* r_lnhh_g = (const float*)d_in[14];
  const float* r_lnhh_b = (const float*)d_in[15];
  const float* r_lnc_g = (const float*)d_in[16];
  const float* r_lnc_b = (const float*)d_in[17];
  const float* r_proj_w = (const float*)d_in[18];
  const float* r_proj_b = (const float*)d_in[19];
  const float* p_proj_w = (const float*)d_in[20];
  const float* p_proj_b = (const float*)d_in[21];
  const float* p_ffd_w = (const float*)d_in[22];
  const float* p_ffd_b = (const float*)d_in[23];
  const float* p_bn_g = (const float*)d_in[24];
  const float* p_bn_b = (const float*)d_in[25];

  float* fbase = (float*)d_ws;
  size_t off = 0;
  auto A = [&](size_t n) {
    float* pp = fbase + off;
    off += (n + 63) & ~(size_t)63;
    return pp;
  };
  float* emb = A((size_t)T_N * B_N * D_N);
  float* h1 = A((size_t)B_N * T_N * H_N);
  float* gate = A(B_N * T_N);
  float* gaten = A(B_N * T_N);
  float* mg = A((size_t)T_N * B_N * S_N);
  float* mgn = A((size_t)T_N * B_N * S_N);
  float* memh = A((size_t)NL_N * B_N * S_N * H_N);
  float* memc = A((size_t)NL_N * B_N * S_N * H_N);
  float* pmem = A((size_t)B_N * S_N * H_N);
  float* outh = A((size_t)T_N * B_N * H_N);
  float* outsel = A((size_t)T_N * B_N * H_N);
  float* keyg = A(B_N * H_N);
  float* selc = A(B_N * H_N);
  float* gih = A(B_N * G4_N);
  float* ghh = A(B_N * G4_N);
  float* hglob = A(B_N * H_N);

  _Float16* hbse = (_Float16*)(fbase + off);
  size_t hoff = 0;
  auto Hh = [&](size_t n) {
    _Float16* pp = hbse + hoff;
    hoff += (n + 127) & ~(size_t)127;
    return pp;
  };
  _Float16* wconv = Hh(72 * 24 * 512);
  _Float16* wproj = Hh(2 * 24 * 24 * 512);
  _Float16* wih = Hh(2 * 12 * 96 * 512);
  _Float16* whh = Hh(2 * 12 * 96 * 512);
  _Float16* wpproj = Hh(24 * 24 * 512);
  _Float16* wffd = Hh(24 * 24 * 512);

  // -- pack weights to f16 WMMA fragments --
  pack_conv_kernel<<<(72 * 24 * 512 + 255) / 256, 256, 0, stream>>>(
      pconv1_w, wconv, 72 * 24 * 512);
  for (int j = 0; j < NL_N; ++j) {
    pack_w_kernel<<<(24 * 24 * 512 + 255) / 256, 256, 0, stream>>>(
        r_proj_w + (size_t)j * H_N * (D_N + H_N),
        wproj + (size_t)j * 24 * 24 * 512, H_N, D_N + H_N, 24 * 24 * 512);
    pack_w_kernel<<<(12 * 96 * 512 + 255) / 256, 256, 0, stream>>>(
        r_wih + (size_t)j * G4_N * D_N, wih + (size_t)j * 12 * 96 * 512, G4_N,
        D_N, 12 * 96 * 512);
    pack_w_kernel<<<(12 * 96 * 512 + 255) / 256, 256, 0, stream>>>(
        r_whh + (size_t)j * G4_N * H_N, whh + (size_t)j * 12 * 96 * 512, G4_N,
        H_N, 12 * 96 * 512);
  }
  pack_w_kernel<<<(24 * 24 * 512 + 255) / 256, 256, 0, stream>>>(
      p_proj_w, wpproj, H_N, 2 * H_N, 24 * 24 * 512);
  pack_w_kernel<<<(24 * 24 * 512 + 255) / 256, 256, 0, stream>>>(
      p_ffd_w, wffd, D_N, 2 * H_N, 24 * 24 * 512);

  // -- embedding + state init --
  embed_kernel<<<(T_N * B_N * D_N + 255) / 256, 256, 0, stream>>>(
      ids, emb_w, emb, T_N * B_N * D_N);
  int zeroN = NL_N * B_N * S_N * H_N * 2 + B_N * S_N * H_N;
  zero_kernel<<<(zeroN + 255) / 256, 256, 0, stream>>>(memh, zeroN);

  // -- parsing network --
  conv_gemm_kernel<<<1152, 256, 0, stream>>>(emb, wconv, pconv1_b, pbn_g, pbn_b,
                                             h1);
  gates_kernel<<<768, 256, 0, stream>>>(h1, pconv2_w, pconv2_b, gate, gaten);
  memgates_kernel<<<(B_N * T_N + 255) / 256, 256, 0, stream>>>(gate, gaten, mg,
                                                               mgn);

  // -- sequential scan --
  SP sp;
  sp.emb = emb; sp.mg = mg; sp.mgn = mgn;
  sp.r_bih = r_bih; sp.r_bhh = r_bhh;
  sp.r_lnih_g = r_lnih_g; sp.r_lnih_b = r_lnih_b;
  sp.r_lnhh_g = r_lnhh_g; sp.r_lnhh_b = r_lnhh_b;
  sp.r_lnc_g = r_lnc_g; sp.r_lnc_b = r_lnc_b;
  sp.r_proj_b = r_proj_b; sp.p_proj_b = p_proj_b;
  sp.wproj = wproj; sp.wih = wih; sp.whh = whh; sp.wpproj = wpproj;
  sp.memh = memh; sp.memc = memc; sp.pmem = pmem;
  sp.outh = outh; sp.outsel = outsel;
  sp.keyg = keyg; sp.selc = selc; sp.gih = gih; sp.ghh = ghh; sp.hglob = hglob;
  for (int t = 0; t < T_N; ++t)
    step_kernel<<<1, 512, 0, stream>>>(sp, t);

  // -- output head + mask --
  head_gemm_kernel<<<1152, 256, 0, stream>>>(outh, outsel, wffd, p_ffd_b, p_bn_g,
                                             p_bn_b, (float*)d_out);
  mask_kernel<<<(B_N * T_N + 255) / 256, 256, 0, stream>>>(
      ids, (float*)d_out + (size_t)T_N * B_N * D_N, B_N * T_N);
}
